// RNN_71880572666275
// MI455X (gfx1250) — matrix operand order repaired and checked by reading, not compiled, not run on hardware
//
#include <hip/hip_runtime.h>
#include <hip/hip_bf16.h>

// GRU over reversed sequence: B=4096, T=82, D=128, H=64, G=3H=192.
// Phase 1: xp[t][b][g] = emb[idx[b][T-1-t]] @ W_ih^T + b_ih (+ b_hh for r/z cols)
//          via v_wmma_f32_16x16x32_bf16, f32 accumulate, inline-0 C init.
// Phase 2: one wave per 16 batch rows runs the whole scan; h stays in registers;
//          W_hh bf16 in LDS; next step's xp tile double-buffered into LDS by the
//          Tensor Data Mover (tensor_load_to_lds + s_wait_tensorcnt).
// Workspace: xp needs T*B*G*4 = 258 MB in d_ws.

#define Bsz 4096
#define Tsz 82
#define Dsz 128
#define Hsz 64
#define Gsz 192

typedef __attribute__((ext_vector_type(16))) __bf16        v16bf;
typedef __attribute__((ext_vector_type(8)))  float         v8f;
typedef __attribute__((ext_vector_type(4)))  unsigned int  u32x4;
typedef __attribute__((ext_vector_type(8)))  int           i32x8;
typedef __attribute__((ext_vector_type(4)))  int           i32x4;

#if __has_builtin(__builtin_amdgcn_tensor_load_to_lds) && \
    __has_builtin(__builtin_amdgcn_s_wait_tensorcnt)
#define GRU_USE_TDM 1
#else
#define GRU_USE_TDM 0
#endif

union FragBF {
    u32x4          q[2];
    unsigned short u[16];
    v16bf          v;
};

__device__ __forceinline__ unsigned short f2bf(float f) {
    union { float f; unsigned int u; } a; a.f = f;
    unsigned int u = a.u;
    u += 0x7fffu + ((u >> 16) & 1u);   // round-to-nearest-even
    return (unsigned short)(u >> 16);
}

__device__ __forceinline__ v16bf ld_frag(const unsigned short* p_lo,
                                         const unsigned short* p_hi) {
    FragBF f;
    f.q[0] = *(const u32x4*)p_lo;
    f.q[1] = *(const u32x4*)p_hi;
    return f.v;
}

__device__ __forceinline__ v16bf cvt_frag(const float* p_lo, const float* p_hi) {
    FragBF f;
#pragma unroll
    for (int i = 0; i < 8; ++i) f.u[i] = f2bf(p_lo[i]);
#pragma unroll
    for (int i = 0; i < 8; ++i) f.u[8 + i] = f2bf(p_hi[i]);
    return f.v;
}

#if GRU_USE_TDM
// TDM: async-load one contiguous run of `nelem` f32 (global -> LDS).
// D# group0: count=1, lds_addr, global_addr, type=2.
// D# group1: data_size=4B, tensor_dim0=tile_dim0=nelem, dims1.. = 1/unused.
__device__ __forceinline__ void tdm_load_f32(const float* gsrc, unsigned lds_off,
                                             int nelem) {
    unsigned long long ga = (unsigned long long)(uintptr_t)gsrc;
    u32x4 g0 = { 1u,                                   // count=1, user mode
                 lds_off,                              // lds_addr (bytes)
                 (unsigned)ga,                         // global_addr[31:0]
                 ((unsigned)(ga >> 32) & 0x01FFFFFFu)  // global_addr[56:32]
                   | 0x80000000u };                    // type=2 ("image")
    i32x8 g1 = { 0x00020000,            // data_size = 4 bytes
                 (int)((unsigned)nelem << 16),  // tensor_dim0[15:0] @ bits[63:48]
                 1 << 16,               // tensor_dim0 hi=0 | tensor_dim1 = 1
                 (int)((unsigned)nelem << 16),  // tensor_dim1 hi=0 | tile_dim0
                 1,                     // tile_dim1 = 1, tile_dim2 = 0
                 nelem,                 // tensor_dim0_stride lo
                 0, 0 };
    i32x4 gz = { 0, 0, 0, 0 };
#if __clang_major__ >= 23
    i32x8 gz8 = { 0, 0, 0, 0, 0, 0, 0, 0 };
    __builtin_amdgcn_tensor_load_to_lds(g0, g1, gz, gz, gz8, 0);
#else
    __builtin_amdgcn_tensor_load_to_lds(g0, g1, gz, gz, 0);
#endif
}
__device__ __forceinline__ unsigned lds_off_of(const void* p) {
    // Generic LDS pointer: low 32 bits are the wave-relative LDS byte offset.
    return (unsigned)(uintptr_t)p;
}
#endif

// ---------------------------------------------------------------------------
// Phase 1: input projection. Grid = (B/128, T), 256 threads = 8 waves.
// ---------------------------------------------------------------------------
__global__ __launch_bounds__(256) void gru_xproj_kernel(
    const int*   __restrict__ idx,   // [B][T]
    const float* __restrict__ emb,   // [V][D]
    const float* __restrict__ W_ih,  // [G][D]
    const float* __restrict__ b_ih,  // [G]
    const float* __restrict__ b_hh,  // [G] (r/z parts folded into xp)
    float*       __restrict__ xp)    // [T][B][G], time reversed
{
    __shared__ unsigned short sW[Gsz * Dsz];   // 48 KB bf16 W_ih

    const int tid  = threadIdx.x;
    const int t    = blockIdx.y;
    const int b0   = blockIdx.x * 128;
    const int tsrc = Tsz - 1 - t;

    for (int i = tid * 4; i < Gsz * Dsz; i += 256 * 4) {
        float4 w = *(const float4*)(W_ih + i);
        sW[i + 0] = f2bf(w.x); sW[i + 1] = f2bf(w.y);
        sW[i + 2] = f2bf(w.z); sW[i + 3] = f2bf(w.w);
    }
    __syncthreads();

    const int wave = tid >> 5;
    const int lane = tid & 31;
    const int r    = lane & 15;
    const int hsel = lane >> 4;
    const int m0   = wave * 16;

    const int    e    = idx[(b0 + m0 + r) * Tsz + tsrc];
    const float* xrow = emb + (long long)e * Dsz;

    // Per-lane bias (depends on column only): b_ih everywhere, + b_hh for r/z.
    float bi[12];
#pragma unroll
    for (int j = 0; j < 12; ++j) {
        bi[j] = b_ih[j * 16 + r] + (j < 8 ? b_hh[j * 16 + r] : 0.0f);
    }

    v8f acc[12];
    const v8f zero = {};

    // kc = 0: inline-constant C=0 (no accumulator-init movs)
    {
        const int ka = hsel * 8;
        v16bf afrag = cvt_frag(xrow + ka, xrow + ka + 16);
        const int kb = hsel * 8;
#pragma unroll
        for (int j = 0; j < 12; ++j) {
            const unsigned short* wp = sW + (j * 16 + r) * Dsz + kb;
            acc[j] = __builtin_amdgcn_wmma_f32_16x16x32_bf16(
                false, afrag, false, ld_frag(wp, wp + 16), (short)0, zero, false, false);
        }
    }
#pragma unroll
    for (int kc = 1; kc < 4; ++kc) {
        const int ka = kc * 32 + hsel * 8;
        v16bf afrag = cvt_frag(xrow + ka, xrow + ka + 16);
        const int kb = kc * 32 + hsel * 8;
#pragma unroll
        for (int j = 0; j < 12; ++j) {
            const unsigned short* wp = sW + (j * 16 + r) * Dsz + kb;
            acc[j] = __builtin_amdgcn_wmma_f32_16x16x32_bf16(
                false, afrag, false, ld_frag(wp, wp + 16), (short)0, acc[j], false, false);
        }
    }

    float* outp = xp + ((long long)t * Bsz + b0 + m0) * Gsz;
#pragma unroll
    for (int j = 0; j < 12; ++j) {
        const int col = j * 16 + r;
#pragma unroll
        for (int v = 0; v < 8; ++v)
            outp[(long long)(hsel * 8 + v) * Gsz + col] = acc[j][v] + bi[j];
    }
}

// ---------------------------------------------------------------------------
// Phase 2: sequential scan. Grid = B/16 = 256 single-wave blocks.
// LDS: 24 KB W_hh + 2 KB h-image + 2 x 12 KB xp double buffer = 50 KB.
// ---------------------------------------------------------------------------
__global__ __launch_bounds__(32) void gru_scan_kernel(
    const float* __restrict__ hidden,  // [2][B][H]; use hidden[1]
    const float* __restrict__ W_hh,    // [G][H]
    const float* __restrict__ b_hh,    // [G] (only n-part used here)
    const float* __restrict__ xp,      // [T][B][G]
    float*       __restrict__ out)     // [B][H]
{
    __shared__ __align__(16) unsigned char smem[24576 + 2048 + 2 * 12288];
    unsigned short* sW = (unsigned short*)smem;                 // [192][64] bf16
    unsigned short* sH = (unsigned short*)(smem + 24576);       // [16][64]  bf16
    float* xb0 = (float*)(smem + 24576 + 2048);                 // [16][192] f32
    float* xb1 = xb0 + 16 * Gsz;

    const int lane = threadIdx.x & 31;
    const int b0   = blockIdx.x * 16;
    const int r    = lane & 15;
    const int hsel = lane >> 4;

    // Stage W_hh -> bf16 LDS (single wave).
    for (int i = lane * 4; i < Gsz * Hsz; i += 32 * 4) {
        float4 w = *(const float4*)(W_hh + i);
        sW[i + 0] = f2bf(w.x); sW[i + 1] = f2bf(w.y);
        sW[i + 2] = f2bf(w.z); sW[i + 3] = f2bf(w.w);
    }

    // n-gate recurrent bias (cannot fold into xp: multiplied by r).
    float bn[4];
#pragma unroll
    for (int j = 0; j < 4; ++j) bn[j] = b_hh[128 + j * 16 + r];

    // h0 = hidden[1], loaded in C layout; seed bf16 A-image.
    const float* h0 = hidden + (long long)Bsz * Hsz;
    v8f h[4];
#pragma unroll
    for (int j = 0; j < 4; ++j)
#pragma unroll
        for (int v = 0; v < 8; ++v) {
            h[j][v] = h0[(long long)(b0 + hsel * 8 + v) * Hsz + j * 16 + r];
            sH[(hsel * 8 + v) * Hsz + j * 16 + r] = f2bf(h[j][v]);
        }

#if GRU_USE_TDM
    // Preload step 0's xp tile (contiguous 16*192 f32) via the TDM.
    tdm_load_f32(xp + (long long)b0 * Gsz, lds_off_of(xb0), 16 * Gsz);
#endif

    const v8f zero = {};
    for (int t = 0; t < Tsz; ++t) {
        v8f acc[12];
        // gh = h @ W_hh^T : K=64, two 32-wide chunks; kc0 uses inline C=0.
        {
            const unsigned short* ap = sH + r * Hsz + hsel * 8;
            v16bf afrag = ld_frag(ap, ap + 16);
            const int kb = hsel * 8;
#pragma unroll
            for (int j = 0; j < 12; ++j) {
                const unsigned short* wp = sW + (j * 16 + r) * Hsz + kb;
                acc[j] = __builtin_amdgcn_wmma_f32_16x16x32_bf16(
                    false, afrag, false, ld_frag(wp, wp + 16), (short)0, zero, false, false);
            }
        }
        {
            const unsigned short* ap = sH + r * Hsz + 32 + hsel * 8;
            v16bf afrag = ld_frag(ap, ap + 16);
            const int kb = 32 + hsel * 8;
#pragma unroll
            for (int j = 0; j < 12; ++j) {
                const unsigned short* wp = sW + (j * 16 + r) * Hsz + kb;
                acc[j] = __builtin_amdgcn_wmma_f32_16x16x32_bf16(
                    false, afrag, false, ld_frag(wp, wp + 16), (short)0, acc[j], false, false);
            }
        }

#if GRU_USE_TDM
        if (t + 1 < Tsz) {   // overlap next tile's DMA with this step's gates
            float* nb = ((t + 1) & 1) ? xb1 : xb0;
            tdm_load_f32(xp + ((long long)(t + 1) * Bsz + b0) * Gsz,
                         lds_off_of(nb), 16 * Gsz);
            __builtin_amdgcn_s_wait_tensorcnt((short)1);
        } else {
            __builtin_amdgcn_s_wait_tensorcnt((short)0);
        }
        asm volatile("" ::: "memory");   // keep LDS reads below the wait
        const float* xpt = (t & 1) ? xb1 : xb0;
#else
        const float* xpt = xp + ((long long)t * Bsz + b0) * Gsz;
        if (t + 1 < Tsz) {
            const float* xpn = xp + ((long long)(t + 1) * Bsz + b0) * Gsz;
            __builtin_prefetch(xpn + (lane << 7), 0, 1);
        }
#endif

        // Gates (r/z biases pre-folded into xp in phase 1).
#pragma unroll
        for (int j = 0; j < 4; ++j) {
#pragma unroll
            for (int v = 0; v < 8; ++v) {
                const int ro = (hsel * 8 + v) * Gsz;
                const float xr = xpt[ro +       j * 16 + r];
                const float xz = xpt[ro +  64 + j * 16 + r];
                const float xn = xpt[ro + 128 + j * 16 + r];
                const float rg = 1.0f / (1.0f + __expf(-(xr + acc[j][v])));
                const float zg = 1.0f / (1.0f + __expf(-(xz + acc[j + 4][v])));
                const float ng = tanhf(xn + rg * (acc[j + 8][v] + bn[j]));
                h[j][v] = (1.0f - zg) * ng + zg * h[j][v];
            }
        }

        // Refresh bf16 h A-image for the next step (wave-private).
#pragma unroll
        for (int j = 0; j < 4; ++j)
#pragma unroll
            for (int v = 0; v < 8; ++v)
                sH[(hsel * 8 + v) * Hsz + j * 16 + r] = f2bf(h[j][v]);
    }

#pragma unroll
    for (int j = 0; j < 4; ++j)
#pragma unroll
        for (int v = 0; v < 8; ++v)
            out[(long long)(b0 + hsel * 8 + v) * Hsz + j * 16 + r] = h[j][v];
}

extern "C" void kernel_launch(void* const* d_in, const int* in_sizes, int n_in,
                              void* d_out, int out_size, void* d_ws, size_t ws_size,
                              hipStream_t stream) {
    (void)in_sizes; (void)n_in; (void)out_size; (void)ws_size;
    const int*   idx    = (const int*)  d_in[0];
    const float* hidden = (const float*)d_in[1];
    const float* emb    = (const float*)d_in[2];
    const float* W_ih   = (const float*)d_in[3];
    const float* W_hh   = (const float*)d_in[4];
    const float* b_ih   = (const float*)d_in[5];
    const float* b_hh   = (const float*)d_in[6];
    float* out = (float*)d_out;
    float* xp  = (float*)d_ws;   // [T][B][3H] f32 (258 MB)

    dim3 g1(Bsz / 128, Tsz);
    gru_xproj_kernel<<<g1, 256, 0, stream>>>(idx, emb, W_ih, b_ih, b_hh, xp);
    gru_scan_kernel<<<Bsz / 16, 32, 0, stream>>>(hidden, W_hh, b_hh, xp, out);
}
